// MeshGraphNetBaseline_88940182765947
// MI455X (gfx1250) — matrix operand order repaired and checked by reading, not compiled, not run on hardware
//
#include <hip/hip_runtime.h>
#include <cstddef>
#include <cstdint>

// ---------------------------------------------------------------------------
// Types for CDNA5 WMMA (wave32): A/B = 16 halves (8 VGPRs), C/D = 8 floats.
// ---------------------------------------------------------------------------
typedef __attribute__((ext_vector_type(16))) _Float16 v16h;
typedef __attribute__((ext_vector_type(8)))  _Float16 v8h;
typedef __attribute__((ext_vector_type(8)))  float    v8f;
typedef __attribute__((ext_vector_type(4)))  float    v4f;

#define N_NODES 100000
#define N_EDGES 1600000
#define HID     64

// Fast SiLU: v_exp_f32 + v_add + v_rcp_f32 + v_mul (TRANS ops co-execute with
// the XDL/WMMA pipe). Avoids the ~12-op IEEE-exact division expansion.
__device__ __forceinline__ float silu_f(float x) {
    return x * __builtin_amdgcn_rcpf(1.0f + __expf(-x));
}

// A-operand (16x32 f16) fragment load. Contiguous k order used here (not the ISA table order):
// lane L holds row M = L&15; halves 0..15 -> K = hi*16 + i (matches pack_b layout)
// where hi = L>>4.  rowBase points at a >=64-half contiguous row; col0 in {0,32}.
__device__ __forceinline__ v16h load_a16(const _Float16* rowBase, int col0, int hi) {
    v8h c1 = *(const v8h*)(rowBase + col0 + hi * 16);
    v8h c2 = *(const v8h*)(rowBase + col0 + hi * 16 + 8);
    v16h a;
#pragma unroll
    for (int i = 0; i < 8; ++i) { a[i] = c1[i]; a[8 + i] = c2[i]; }
    return a;
}

// ---------------------------------------------------------------------------
// Weight pre-swizzle into B-operand (KxN f16, K%32==0, N==64) lane layout:
// packed[((nt*KC + kc)*32 + lane)*16 + e]  with
//   nt = n/16, kc = k/32, lane = ((k>>4)&1)*16 + (n&15), e = k&15
// so each lane's 16-half B fragment is one contiguous 32B chunk.
// ---------------------------------------------------------------------------
__global__ void pack_b_kernel(const float* __restrict__ W, _Float16* __restrict__ P,
                              int K, int Nn) {
    int i = blockIdx.x * blockDim.x + threadIdx.x;
    if (i >= K * Nn) return;
    int k = i / Nn, n = i % Nn;
    int nt = n >> 4, kc = k >> 5;
    int lane = (((k >> 4) & 1) << 4) | (n & 15);
    int e = k & 15;
    int KC = K >> 5;
    P[(((size_t)nt * KC + kc) * 32 + lane) * 16 + e] = (_Float16)W[i];
}

// h = node_feats @ in_w + in_b   (N,3)x(3,64)
__global__ void inproj_kernel(const float* __restrict__ nf, const float* __restrict__ w,
                              const float* __restrict__ b, float* __restrict__ h, int total) {
    int i = blockIdx.x * blockDim.x + threadIdx.x;
    if (i >= total) return;
    int node = i >> 6, c = i & 63;
    float s = b[c];
#pragma unroll
    for (int j = 0; j < 3; ++j) s += nf[node * 3 + j] * w[j * 64 + c];
    h[i] = s;
}

// h16 = f16(h); agg = 0   (8 elements / thread, b128 traffic)
__global__ void conv_zero_kernel(const float* __restrict__ h, _Float16* __restrict__ h16,
                                 float* __restrict__ agg, int total8) {
    int i = blockIdx.x * blockDim.x + threadIdx.x;
    if (i >= total8) return;
    v4f a = ((const v4f*)h)[2 * i];
    v4f c = ((const v4f*)h)[2 * i + 1];
    v8h o;
#pragma unroll
    for (int j = 0; j < 4; ++j) { o[j] = (_Float16)a[j]; o[4 + j] = (_Float16)c[j]; }
    ((v8h*)h16)[i] = o;
    v4f z = {0.f, 0.f, 0.f, 0.f};
    ((v4f*)agg)[2 * i] = z;
    ((v4f*)agg)[2 * i + 1] = z;
}

__global__ void cvt_kernel(const float* __restrict__ x, _Float16* __restrict__ y, int total8) {
    int i = blockIdx.x * blockDim.x + threadIdx.x;
    if (i >= total8) return;
    v4f a = ((const v4f*)x)[2 * i];
    v4f c = ((const v4f*)x)[2 * i + 1];
    v8h o;
#pragma unroll
    for (int j = 0; j < 4; ++j) { o[j] = (_Float16)a[j]; o[4 + j] = (_Float16)c[j]; }
    ((v8h*)y)[i] = o;
}

// ---------------------------------------------------------------------------
// Edge message MLP:  m = silu([h_s,h_d] @ W1 + b1) @ W2 + b2 ; agg[dst] += m
// One wave per 16-edge tile, grid-stride. Weights held in registers.
// ---------------------------------------------------------------------------
__global__ void __launch_bounds__(256)
msg_kernel(const _Float16* __restrict__ h16, const int* __restrict__ ei,
           const v16h* __restrict__ w1p, const float* __restrict__ b1,
           const v16h* __restrict__ w2p, const float* __restrict__ b2,
           float* __restrict__ agg, int nTiles, int E) {
    __shared__ _Float16 lds[8][16][72];  // per-wave 16x64 activation tile (+pad)

    const int lane = threadIdx.x & 31;
    const int wv   = threadIdx.x >> 5;
    const int col  = lane & 15;
    const int hi   = lane >> 4;
    const int gw   = blockIdx.x * (blockDim.x >> 5) + wv;
    const int nw   = gridDim.x * (blockDim.x >> 5);

    // B fragments in registers: W1 4 N-tiles x 4 K-steps, W2 4 x 2.
    v16h w1r[4][4], w2r[4][2];
    float b1v[4], b2v[4];
#pragma unroll
    for (int nt = 0; nt < 4; ++nt) {
#pragma unroll
        for (int kc = 0; kc < 4; ++kc) w1r[nt][kc] = w1p[(nt * 4 + kc) * 32 + lane];
#pragma unroll
        for (int kc = 0; kc < 2; ++kc) w2r[nt][kc] = w2p[(nt * 2 + kc) * 32 + lane];
        b1v[nt] = b1[nt * 16 + col];
        b2v[nt] = b2[nt * 16 + col];
    }

    for (int t = gw; t < nTiles; t += nw) {
        const int base = t * 16;
        const int er = base + col;
        const int sN = ei[er];
        const int dN = ei[E + er];

        // A fragments for K=128: [h_src(0..63) | h_dst(64..127)]
        v16h a0 = load_a16(h16 + (size_t)sN * 64, 0, hi);
        v16h a1 = load_a16(h16 + (size_t)sN * 64, 32, hi);
        v16h a2 = load_a16(h16 + (size_t)dN * 64, 0, hi);
        v16h a3 = load_a16(h16 + (size_t)dN * 64, 32, hi);

        __builtin_amdgcn_wave_barrier();
#pragma unroll
        for (int nt = 0; nt < 4; ++nt) {
            v8f acc;
#pragma unroll
            for (int i = 0; i < 8; ++i) acc[i] = b1v[nt];
            acc = __builtin_amdgcn_wmma_f32_16x16x32_f16(false, a0, false, w1r[nt][0], (short)0, acc, false, false);
            acc = __builtin_amdgcn_wmma_f32_16x16x32_f16(false, a1, false, w1r[nt][1], (short)0, acc, false, false);
            acc = __builtin_amdgcn_wmma_f32_16x16x32_f16(false, a2, false, w1r[nt][2], (short)0, acc, false, false);
            acc = __builtin_amdgcn_wmma_f32_16x16x32_f16(false, a3, false, w1r[nt][3], (short)0, acc, false, false);
            // SiLU, D-layout (lane=col, vgpr=row) -> LDS row-major
#pragma unroll
            for (int r = 0; r < 8; ++r)
                lds[wv][r + hi * 8][nt * 16 + col] = (_Float16)silu_f(acc[r]);
        }
        __builtin_amdgcn_wave_barrier();
        asm volatile("s_wait_dscnt 0" ::: "memory");  // wave-private LDS RAW

        // Reload as A-layout (row = lane&15) for GEMM2 (K=64)
        const _Float16* lrow = &lds[wv][col][0];
        v16h x0 = load_a16(lrow, 0, hi);
        v16h x1 = load_a16(lrow, 32, hi);
        __builtin_amdgcn_wave_barrier();

        int dsts[8];
#pragma unroll
        for (int r = 0; r < 8; ++r) dsts[r] = ei[E + base + r + hi * 8];

#pragma unroll
        for (int nt = 0; nt < 4; ++nt) {
            v8f acc2;
#pragma unroll
            for (int i = 0; i < 8; ++i) acc2[i] = b2v[nt];
            acc2 = __builtin_amdgcn_wmma_f32_16x16x32_f16(false, x0, false, w2r[nt][0], (short)0, acc2, false, false);
            acc2 = __builtin_amdgcn_wmma_f32_16x16x32_f16(false, x1, false, w2r[nt][1], (short)0, acc2, false, false);
#pragma unroll
            for (int r = 0; r < 8; ++r)
                atomicAdd(&agg[(size_t)dsts[r] * 64 + nt * 16 + col], acc2[r]);
        }
    }
}

// ---------------------------------------------------------------------------
// Node update MLP: h += silu([h,agg] @ U1 + ub1) @ U2 + ub2
// ---------------------------------------------------------------------------
__global__ void __launch_bounds__(256)
upd_kernel(float* __restrict__ h, const _Float16* __restrict__ h16,
           const _Float16* __restrict__ agg16,
           const v16h* __restrict__ w1p, const float* __restrict__ b1,
           const v16h* __restrict__ w2p, const float* __restrict__ b2, int nTiles) {
    __shared__ _Float16 lds[8][16][72];

    const int lane = threadIdx.x & 31;
    const int wv   = threadIdx.x >> 5;
    const int col  = lane & 15;
    const int hi   = lane >> 4;
    const int gw   = blockIdx.x * (blockDim.x >> 5) + wv;
    const int nw   = gridDim.x * (blockDim.x >> 5);

    v16h w1r[4][4], w2r[4][2];
    float b1v[4], b2v[4];
#pragma unroll
    for (int nt = 0; nt < 4; ++nt) {
#pragma unroll
        for (int kc = 0; kc < 4; ++kc) w1r[nt][kc] = w1p[(nt * 4 + kc) * 32 + lane];
#pragma unroll
        for (int kc = 0; kc < 2; ++kc) w2r[nt][kc] = w2p[(nt * 2 + kc) * 32 + lane];
        b1v[nt] = b1[nt * 16 + col];
        b2v[nt] = b2[nt * 16 + col];
    }

    for (int t = gw; t < nTiles; t += nw) {
        const int base = t * 16;
        const int node = base + col;
        const _Float16* hrow = h16 + (size_t)node * 64;
        const _Float16* arow = agg16 + (size_t)node * 64;
        v16h a0 = load_a16(hrow, 0, hi);
        v16h a1 = load_a16(hrow, 32, hi);
        v16h a2 = load_a16(arow, 0, hi);
        v16h a3 = load_a16(arow, 32, hi);

        __builtin_amdgcn_wave_barrier();
#pragma unroll
        for (int nt = 0; nt < 4; ++nt) {
            v8f acc;
#pragma unroll
            for (int i = 0; i < 8; ++i) acc[i] = b1v[nt];
            acc = __builtin_amdgcn_wmma_f32_16x16x32_f16(false, a0, false, w1r[nt][0], (short)0, acc, false, false);
            acc = __builtin_amdgcn_wmma_f32_16x16x32_f16(false, a1, false, w1r[nt][1], (short)0, acc, false, false);
            acc = __builtin_amdgcn_wmma_f32_16x16x32_f16(false, a2, false, w1r[nt][2], (short)0, acc, false, false);
            acc = __builtin_amdgcn_wmma_f32_16x16x32_f16(false, a3, false, w1r[nt][3], (short)0, acc, false, false);
#pragma unroll
            for (int r = 0; r < 8; ++r)
                lds[wv][r + hi * 8][nt * 16 + col] = (_Float16)silu_f(acc[r]);
        }
        __builtin_amdgcn_wave_barrier();
        asm volatile("s_wait_dscnt 0" ::: "memory");

        const _Float16* lrow = &lds[wv][col][0];
        v16h x0 = load_a16(lrow, 0, hi);
        v16h x1 = load_a16(lrow, 32, hi);
        __builtin_amdgcn_wave_barrier();

#pragma unroll
        for (int nt = 0; nt < 4; ++nt) {
            v8f acc2;
#pragma unroll
            for (int i = 0; i < 8; ++i) acc2[i] = b2v[nt];
            acc2 = __builtin_amdgcn_wmma_f32_16x16x32_f16(false, x0, false, w2r[nt][0], (short)0, acc2, false, false);
            acc2 = __builtin_amdgcn_wmma_f32_16x16x32_f16(false, x1, false, w2r[nt][1], (short)0, acc2, false, false);
#pragma unroll
            for (int r = 0; r < 8; ++r) {
                int nd = base + r + hi * 8;
                size_t idx = (size_t)nd * 64 + nt * 16 + col;
                h[idx] += acc2[r];  // residual
            }
        }
    }
}

// readout: logits[i] = silu(h_i @ W1 + b1) @ w2 + b2
__global__ void readout_kernel(const float* __restrict__ h, const float* __restrict__ w1,
                               const float* __restrict__ b1, const float* __restrict__ w2,
                               const float* __restrict__ b2, float* __restrict__ logits, int n) {
    int i = blockIdx.x * blockDim.x + threadIdx.x;
    if (i >= n) return;
    const float* hr = h + (size_t)i * 64;
    float hrl[64];
#pragma unroll
    for (int k = 0; k < 64; ++k) hrl[k] = hr[k];
    float lg = b2[0];
#pragma unroll 1
    for (int j = 0; j < 64; ++j) {
        float acc = b1[j];
#pragma unroll
        for (int k = 0; k < 64; ++k) acc += hrl[k] * w1[k * 64 + j];
        lg += silu_f(acc) * w2[j];
    }
    logits[i] = lg;
}

// ------------------------- softmax over N (3 passes) -----------------------
__global__ void redmax_kernel(const float* __restrict__ x, int n, float* __restrict__ part) {
    __shared__ float sm[256];
    int tid = threadIdx.x;
    float m = -3.4e38f;
    for (int i = blockIdx.x * 256 + tid; i < n; i += gridDim.x * 256) m = fmaxf(m, x[i]);
    sm[tid] = m;
    __syncthreads();
    for (int s = 128; s > 0; s >>= 1) {
        if (tid < s) sm[tid] = fmaxf(sm[tid], sm[tid + s]);
        __syncthreads();
    }
    if (tid == 0) part[blockIdx.x] = sm[0];
}

__global__ void finalmax_kernel(const float* __restrict__ part, float* __restrict__ scal) {
    __shared__ float sm[256];
    int tid = threadIdx.x;
    sm[tid] = part[tid];
    __syncthreads();
    for (int s = 128; s > 0; s >>= 1) {
        if (tid < s) sm[tid] = fmaxf(sm[tid], sm[tid + s]);
        __syncthreads();
    }
    if (tid == 0) scal[0] = sm[0];
}

__global__ void exp_kernel(const float* __restrict__ x, int n, const float* __restrict__ scal,
                           float* __restrict__ out, float* __restrict__ part) {
    __shared__ float sm[256];
    int tid = threadIdx.x;
    float mx = scal[0];
    float s = 0.0f;
    for (int i = blockIdx.x * 256 + tid; i < n; i += gridDim.x * 256) {
        float v = __expf(x[i] - mx);
        out[i] = v;
        s += v;
    }
    sm[tid] = s;
    __syncthreads();
    for (int st = 128; st > 0; st >>= 1) {
        if (tid < st) sm[tid] += sm[tid + st];
        __syncthreads();
    }
    if (tid == 0) part[blockIdx.x] = sm[0];
}

__global__ void finalsum_kernel(const float* __restrict__ part, float* __restrict__ scal) {
    __shared__ float sm[256];
    int tid = threadIdx.x;
    sm[tid] = part[tid];
    __syncthreads();
    for (int s = 128; s > 0; s >>= 1) {
        if (tid < s) sm[tid] += sm[tid + s];
        __syncthreads();
    }
    if (tid == 0) scal[1] = sm[0];
}

__global__ void norm_kernel(float* __restrict__ out, int n, const float* __restrict__ scal) {
    int i = blockIdx.x * blockDim.x + threadIdx.x;
    if (i >= n) return;
    out[i] = out[i] / scal[1];
}

// ---------------------------------------------------------------------------
extern "C" void kernel_launch(void* const* d_in, const int* in_sizes, int n_in,
                              void* d_out, int out_size, void* d_ws, size_t ws_size,
                              hipStream_t stream) {
    (void)in_sizes; (void)n_in; (void)out_size; (void)ws_size;
    const float* node_feats = (const float*)d_in[0];
    const int*   edge_index = (const int*)d_in[1];
    const float* in_w  = (const float*)d_in[2];
    const float* in_b  = (const float*)d_in[3];
    const float* msg_w1 = (const float*)d_in[4];
    const float* msg_b1 = (const float*)d_in[5];
    const float* msg_w2 = (const float*)d_in[6];
    const float* msg_b2 = (const float*)d_in[7];
    const float* upd_w1 = (const float*)d_in[8];
    const float* upd_b1 = (const float*)d_in[9];
    const float* upd_w2 = (const float*)d_in[10];
    const float* upd_b2 = (const float*)d_in[11];
    const float* ro_w1 = (const float*)d_in[12];
    const float* ro_b1 = (const float*)d_in[13];
    const float* ro_w2 = (const float*)d_in[14];
    const float* ro_b2 = (const float*)d_in[15];
    float* out = (float*)d_out;

    const int N = N_NODES, E = N_EDGES, total = N * HID;
    const int total8 = total / 8;

    // workspace carve (256B aligned)
    size_t off = 0;
    auto alloc = [&](size_t bytes) -> void* {
        void* p = (char*)d_ws + off;
        off += (bytes + 255) & ~(size_t)255;
        return p;
    };
    float*     h      = (float*)alloc((size_t)total * 4);
    float*     agg    = (float*)alloc((size_t)total * 4);
    _Float16*  h16    = (_Float16*)alloc((size_t)total * 2);
    _Float16*  agg16  = (_Float16*)alloc((size_t)total * 2);
    _Float16*  pbase  = (_Float16*)alloc((size_t)3 * 24576 * 2);
    float*     logits = (float*)alloc((size_t)N * 4);
    float*     part   = (float*)alloc(256 * 4);
    float*     scal   = (float*)alloc(8);

    // --- pre-pack all weights into B-operand f16 layout ---
    for (int l = 0; l < 3; ++l) {
        _Float16* pm1 = pbase + (size_t)l * 24576;
        _Float16* pm2 = pm1 + 8192;
        _Float16* pu1 = pm2 + 4096;
        _Float16* pu2 = pu1 + 8192;
        pack_b_kernel<<<(128 * 64 + 255) / 256, 256, 0, stream>>>(msg_w1 + (size_t)l * 8192, pm1, 128, 64);
        pack_b_kernel<<<(64 * 64 + 255) / 256, 256, 0, stream>>>(msg_w2 + (size_t)l * 4096, pm2, 64, 64);
        pack_b_kernel<<<(128 * 64 + 255) / 256, 256, 0, stream>>>(upd_w1 + (size_t)l * 8192, pu1, 128, 64);
        pack_b_kernel<<<(64 * 64 + 255) / 256, 256, 0, stream>>>(upd_w2 + (size_t)l * 4096, pu2, 64, 64);
    }

    // --- input projection ---
    inproj_kernel<<<(total + 255) / 256, 256, 0, stream>>>(node_feats, in_w, in_b, h, total);

    const int eTiles = E / 16;   // 100000
    const int nTiles = N / 16;   // 6250

    for (int l = 0; l < 3; ++l) {
        const v16h* pm1 = (const v16h*)(pbase + (size_t)l * 24576);
        const v16h* pm2 = (const v16h*)(pbase + (size_t)l * 24576 + 8192);
        const v16h* pu1 = (const v16h*)(pbase + (size_t)l * 24576 + 12288);
        const v16h* pu2 = (const v16h*)(pbase + (size_t)l * 24576 + 20480);

        conv_zero_kernel<<<(total8 + 255) / 256, 256, 0, stream>>>(h, h16, agg, total8);
        msg_kernel<<<512, 256, 0, stream>>>(h16, edge_index, pm1, msg_b1 + l * 64,
                                            pm2, msg_b2 + l * 64, agg, eTiles, E);
        cvt_kernel<<<(total8 + 255) / 256, 256, 0, stream>>>(agg, agg16, total8);
        upd_kernel<<<128, 256, 0, stream>>>(h, h16, agg16, pu1, upd_b1 + l * 64,
                                            pu2, upd_b2 + l * 64, nTiles);
    }

    readout_kernel<<<(N + 255) / 256, 256, 0, stream>>>(h, ro_w1, ro_b1, ro_w2, ro_b2, logits, N);

    redmax_kernel<<<256, 256, 0, stream>>>(logits, N, part);
    finalmax_kernel<<<1, 256, 0, stream>>>(part, scal);
    exp_kernel<<<256, 256, 0, stream>>>(logits, N, scal, out, part);
    finalsum_kernel<<<1, 256, 0, stream>>>(part, scal);
    norm_kernel<<<(N + 255) / 256, 256, 0, stream>>>(out, N, scal);
}
